// PointCloudDiffusionModel_26319559590075
// MI455X (gfx1250) — matrix operand order repaired and checked by reading, not compiled, not run on hardware
//
#include <hip/hip_runtime.h>

// ---------------------------------------------------------------------------
// MI455X (gfx1250, wave32) fused point-cloud diffusion forward.
// Compute-bound (~216 GFLOP, ~10 MB traffic) -> f16 WMMA w/ f32 accum.
// v2: 64-row M tile (dynamic LDS, 100KB), 4-way N-blocked wave tasks (one A
// fragment feeds 4 WMMAs), fragment-major packed weights so each lane's B
// fragment is 32 contiguous bytes (-> global_load_b128 pairs).
// ---------------------------------------------------------------------------

typedef __attribute__((ext_vector_type(16))) _Float16 v16h;
typedef __attribute__((ext_vector_type(8)))  float    v8f;

#define NPTS      120000
#define TARGET    30000
#define TABLE     (1 << 18)
#define NBLK_IN   469            // ceil(120000/256)
#define NBLK_GATH 118            // ceil(30000/256)
#define NBLK_STY  235            // ceil(30000/128)
#define MTILE     64
#define NTILES_M  469            // ceil(30000/64)
#define XSTRIDE   264            // 256 + 8 halves pad (bank spread)
#define HSTRIDE   520            // 512 + 8 halves pad
#define MAIN_LDS_BYTES ((MTILE * XSTRIDE + MTILE * HSTRIDE) * 2)   // 100352

// ----- workspace layout (bytes) --------------------------------------------
#define OFF_PE_W2   (size_t)0          // (128/2)*256 u32  = 65536 B
#define OFF_PE_W3   (size_t)65536      // (256/2)*256 u32  = 131072 B
#define OFF_LW1     (size_t)196608     // (1536/2)*512 u32 = 1572864 B
#define OFF_LW2     (size_t)1769472    // (3072/2)*256 u32 = 1572864 B
#define OFF_OW1     (size_t)3342336    // 131072 B
#define OFF_OW2     (size_t)3473408    // 65536 B
#define OFF_MM_PART (size_t)3538944    // 4*64*8 f32
#define OFF_MM      (size_t)3547136    // 4*8 f32 (mn.xyz, vs)
#define OFF_HASH    (size_t)3547392    // 4*TABLE i32
#define OFF_SEL     (size_t)7741696    // 4*120000 i32
#define OFF_BSUM    (size_t)9661696    // 4*512 i32
#define OFF_BEXC    (size_t)9669888    // 4*512 i32
#define OFF_TOT     (size_t)9678080    // 4 i32 (pad 256)
#define OFF_FINAL   (size_t)9678336    // 4*30000 i32
#define OFF_DOWN    (size_t)10158336   // 4*30000*3 f32  (bc: 0,1=noisy b0,b1; 2,3=cond)
#define OFF_GPART   (size_t)11598336   // 2*235*256 f32
#define OFF_SF      (size_t)12079616   // 2*256 f32
#define OFF_TESF    (size_t)12081664   // 2*256 f32
// total ~11.6 MB of d_ws

union PairU { _Float16 h[2]; unsigned int u; };
union Frag  { v16h v; unsigned int u[8]; };

// ===========================================================================
// Weight packing: f32 [K][N] -> f16 fragment-major.
// Fragment (kstep, nt) = the 32x16 B tile for one WMMA; stored as 32 lanes x
// 8 contiguous dwords (lane's dwords = its whole B fragment -> b128 loads).
// dword q of lane (lhi,lm) in fragment = halves (k,k+1), k = kstep*32+(q>>2)*16+lhi*8+2*(q&3),
// n = nt*16+lm.
// ===========================================================================
__global__ __launch_bounds__(256) void pack_weights_kernel(
    const float* __restrict__ src, unsigned int* __restrict__ dst, int K, int N)
{
    int idx = blockIdx.x * 256 + threadIdx.x;
    int total = (K >> 1) * N;
    if (idx >= total) return;
    int q    = idx & 7;
    int lane = (idx >> 3) & 31;
    int frag = idx >> 8;                 // kstep * (N/16) + nt
    int ntiles = N >> 4;
    int kstep = frag / ntiles;
    int nt    = frag - kstep * ntiles;
    int lhi = lane >> 4, lm = lane & 15;
    int k = (kstep << 5) + ((q >> 2) << 4) + (lhi << 3) + ((q & 3) << 1);
    int n = (nt << 4) + lm;
    PairU p;
    p.h[0] = (_Float16)src[(size_t)k * N + n];
    p.h[1] = (_Float16)src[(size_t)(k + 1) * N + n];
    dst[idx] = p.u;
}

// ===========================================================================
// Voxel downsample (deterministic approximation of the reference)
// ===========================================================================
__global__ __launch_bounds__(256) void minmax_partial_kernel(
    const float* __restrict__ noisy, const float* __restrict__ cond,
    float* __restrict__ part)
{
    int bc = blockIdx.y, chunk = blockIdx.x, tid = threadIdx.x;
    const float* pts = ((bc >> 1) ? cond : noisy) + (size_t)(bc & 1) * NPTS * 3;
    __shared__ float smn[256 * 3], smx[256 * 3];
    float mn[3] = {1e30f, 1e30f, 1e30f}, mx[3] = {-1e30f, -1e30f, -1e30f};
    int beg = chunk * 1875, end = beg + 1875;
    if (end > NPTS) end = NPTS;
    for (int i = beg + tid; i < end; i += 256) {
        const float* p = pts + (size_t)i * 3;
        for (int d = 0; d < 3; ++d) {
            float v = p[d];
            mn[d] = fminf(mn[d], v); mx[d] = fmaxf(mx[d], v);
        }
    }
    for (int d = 0; d < 3; ++d) { smn[tid * 3 + d] = mn[d]; smx[tid * 3 + d] = mx[d]; }
    __syncthreads();
    for (int s = 128; s > 0; s >>= 1) {
        if (tid < s)
            for (int d = 0; d < 3; ++d) {
                smn[tid*3+d] = fminf(smn[tid*3+d], smn[(tid+s)*3+d]);
                smx[tid*3+d] = fmaxf(smx[tid*3+d], smx[(tid+s)*3+d]);
            }
        __syncthreads();
    }
    if (tid == 0) {
        float* o = part + ((size_t)bc * 64 + chunk) * 8;
        for (int d = 0; d < 3; ++d) { o[d] = smn[d]; o[3 + d] = smx[d]; }
    }
}

__global__ __launch_bounds__(64) void minmax_final_kernel(
    const float* __restrict__ part, float* __restrict__ mm)
{
    int bc = blockIdx.x, tid = threadIdx.x;
    __shared__ float smn[64 * 3], smx[64 * 3];
    const float* s = part + ((size_t)bc * 64 + tid) * 8;
    for (int d = 0; d < 3; ++d) { smn[tid*3+d] = s[d]; smx[tid*3+d] = s[3+d]; }
    __syncthreads();
    for (int st = 32; st > 0; st >>= 1) {
        if (tid < st)
            for (int d = 0; d < 3; ++d) {
                smn[tid*3+d] = fminf(smn[tid*3+d], smn[(tid+st)*3+d]);
                smx[tid*3+d] = fmaxf(smx[tid*3+d], smx[(tid+st)*3+d]);
            }
        __syncthreads();
    }
    if (tid == 0) {
        float rng[3];
        for (int d = 0; d < 3; ++d) {
            float r = smx[d] - smn[d];
            rng[d] = (r < 1e-6f) ? 1.0f : r;
        }
        float vs = cbrtf(rng[0] * rng[1] * rng[2] / (float)TARGET) * 1.2f;
        if (vs < 1e-6f) vs = 0.001f;
        float* o = mm + (size_t)bc * 8;
        o[0] = smn[0]; o[1] = smn[1]; o[2] = smn[2]; o[3] = vs;
    }
}

__global__ __launch_bounds__(256) void hash_init_kernel(int* __restrict__ tab)
{
    size_t i = (size_t)blockIdx.y * TABLE + blockIdx.x * 256 + threadIdx.x;
    tab[i] = 0x7fffffff;
}

__device__ __forceinline__ unsigned voxel_hash(const float* p, const float* mmrow)
{
    float inv = 1.0f / mmrow[3];
    int vx = (int)floorf((p[0] - mmrow[0]) * inv);
    int vy = (int)floorf((p[1] - mmrow[1]) * inv);
    int vz = (int)floorf((p[2] - mmrow[2]) * inv);
    unsigned h = ((unsigned)vx * 73856093u) ^ ((unsigned)vy * 19349663u) ^
                 ((unsigned)vz * 83492791u);
    return h & (TABLE - 1);
}

__global__ __launch_bounds__(256) void hash_insert_kernel(
    const float* __restrict__ noisy, const float* __restrict__ cond,
    const float* __restrict__ mm, int* __restrict__ tab)
{
    int bc = blockIdx.y;
    int i = blockIdx.x * 256 + threadIdx.x;
    if (i >= NPTS) return;
    const float* pts = ((bc >> 1) ? cond : noisy) + (size_t)(bc & 1) * NPTS * 3;
    unsigned b = voxel_hash(pts + (size_t)i * 3, mm + (size_t)bc * 8);
    atomicMin(&tab[(size_t)bc * TABLE + b], i);   // order-independent -> deterministic
}

__global__ __launch_bounds__(256) void mark_count_kernel(
    const float* __restrict__ noisy, const float* __restrict__ cond,
    const float* __restrict__ mm, const int* __restrict__ tab,
    int* __restrict__ sel, int* __restrict__ bsum)
{
    int bc = blockIdx.y, blk = blockIdx.x, tid = threadIdx.x;
    int i = blk * 256 + tid;
    int sv = 0;
    if (i < NPTS) {
        const float* pts = ((bc >> 1) ? cond : noisy) + (size_t)(bc & 1) * NPTS * 3;
        unsigned b = voxel_hash(pts + (size_t)i * 3, mm + (size_t)bc * 8);
        sv = (tab[(size_t)bc * TABLE + b] == i) ? 1 : 0;
        sel[(size_t)bc * NPTS + i] = sv;
    }
    __shared__ int red[256];
    red[tid] = sv;
    __syncthreads();
    for (int s = 128; s > 0; s >>= 1) {
        if (tid < s) red[tid] += red[tid + s];
        __syncthreads();
    }
    if (tid == 0) bsum[(size_t)bc * 512 + blk] = red[0];
}

__global__ __launch_bounds__(32) void scan_blocks_kernel(
    const int* __restrict__ bsum, int* __restrict__ bexc, int* __restrict__ tot)
{
    int bc = blockIdx.x;
    if (threadIdx.x == 0) {
        int run = 0;
        for (int b = 0; b < NBLK_IN; ++b) {
            bexc[(size_t)bc * 512 + b] = run;
            run += bsum[(size_t)bc * 512 + b];
        }
        tot[bc] = run;
    }
}

__global__ __launch_bounds__(256) void scatter_kernel(
    const int* __restrict__ sel, const int* __restrict__ bexc,
    const int* __restrict__ tot, int* __restrict__ finalIdx)
{
    int bc = blockIdx.y, blk = blockIdx.x, tid = threadIdx.x;
    int i = blk * 256 + tid;
    int valid = (i < NPTS);
    int sv = valid ? sel[(size_t)bc * NPTS + i] : 0;
    __shared__ int sc[256];
    sc[tid] = sv;
    __syncthreads();
    for (int off = 1; off < 256; off <<= 1) {
        int v = (tid >= off) ? sc[tid - off] : 0;
        __syncthreads();
        sc[tid] += v;
        __syncthreads();
    }
    int incl = sc[tid], excl = incl - sv;
    int bex = bexc[(size_t)bc * 512 + blk];
    if (!valid) return;
    if (sv) {
        int pos = bex + excl;                       // reps by ascending index
        if (pos < TARGET) finalIdx[(size_t)bc * TARGET + pos] = i;
    } else {
        int rank = i - (bex + incl);                // fill: lowest unselected
        int pos = tot[bc] + rank;
        if (pos >= 0 && pos < TARGET) finalIdx[(size_t)bc * TARGET + pos] = i;
    }
}

__global__ __launch_bounds__(256) void gather_kernel(
    const float* __restrict__ noisy, const float* __restrict__ cond,
    const int* __restrict__ finalIdx, float* __restrict__ down,
    int* __restrict__ idxOut)
{
    int bc = blockIdx.y;
    int j = blockIdx.x * 256 + threadIdx.x;
    if (j >= TARGET) return;
    int src = finalIdx[(size_t)bc * TARGET + j];
    const float* pts = ((bc >> 1) ? cond : noisy) + (size_t)(bc & 1) * NPTS * 3;
    const float* p = pts + (size_t)src * 3;
    float* o = down + ((size_t)bc * TARGET + j) * 3;
    o[0] = p[0]; o[1] = p[1]; o[2] = p[2];
    if (bc < 2) idxOut[(size_t)bc * TARGET + j] = src;  // noisy cloud -> idx output
}

// ===========================================================================
// Style encoder: per-point MLP (3->64->128) staged in LDS f16, then 128->256
// matmul fused with max-pool over the block's points -> partial maxes.
// ===========================================================================
__global__ __launch_bounds__(256) void style_point_kernel(
    const float* __restrict__ down,
    const float* __restrict__ w1, const float* __restrict__ b1,
    const float* __restrict__ w2, const float* __restrict__ b2,
    const float* __restrict__ w3, const float* __restrict__ b3,
    float* __restrict__ gpart)
{
    int b = blockIdx.y, blk = blockIdx.x, tid = threadIdx.x;
    int cnt = TARGET - blk * 128; if (cnt > 128) cnt = 128;
    __shared__ _Float16 h2s[128][132];
    if (tid < 128 && tid < cnt) {
        int j = blk * 128 + tid;
        const float* p = down + (((size_t)(2 + b)) * TARGET + j) * 3;
        float px = p[0], py = p[1], pz = p[2];
        float h1[64];
        #pragma unroll
        for (int f = 0; f < 64; ++f)
            h1[f] = fmaxf(px * w1[f] + py * w1[64 + f] + pz * w1[128 + f] + b1[f], 0.f);
        for (int k = 0; k < 128; ++k) {
            float a = b2[k];
            #pragma unroll
            for (int f = 0; f < 64; ++f) a += h1[f] * w2[f * 128 + k];
            h2s[tid][k] = (_Float16)fmaxf(a, 0.f);
        }
    }
    __syncthreads();
    int n = tid;
    float mv = -1e30f;
    for (int p = 0; p < cnt; ++p) {
        float d = b3[n];
        for (int k = 0; k < 128; ++k) d += (float)h2s[p][k] * w3[k * 256 + n];
        mv = fmaxf(mv, d);
    }
    gpart[((size_t)b * NBLK_STY + blk) * 256 + n] = mv;
}

__global__ __launch_bounds__(256) void style_mlp_kernel(
    const float* __restrict__ gpart,
    const float* __restrict__ sm_w1, const float* __restrict__ sm_b1,
    const float* __restrict__ sm_w2, const float* __restrict__ sm_b2,
    const float* __restrict__ sp_w,  const float* __restrict__ sp_b,
    float* __restrict__ sf)
{
    int b = blockIdx.x, tid = threadIdx.x;
    __shared__ float g[256], s1[512], s2[256];
    float mv = -1e30f;
    for (int blk = 0; blk < NBLK_STY; ++blk)
        mv = fmaxf(mv, gpart[((size_t)b * NBLK_STY + blk) * 256 + tid]);
    g[tid] = mv;
    __syncthreads();
    for (int r = 0; r < 2; ++r) {
        int n = tid + r * 256;
        float a = sm_b1[n];
        for (int k = 0; k < 256; ++k) a += g[k] * sm_w1[(size_t)k * 512 + n];
        s1[n] = fmaxf(a, 0.f);
    }
    __syncthreads();
    { float a = sm_b2[tid];
      for (int k = 0; k < 512; ++k) a += s1[k] * sm_w2[(size_t)k * 256 + tid];
      s2[tid] = fmaxf(a, 0.f); }
    __syncthreads();
    { float a = sp_b[tid];
      for (int k = 0; k < 256; ++k) a += s2[k] * sp_w[(size_t)k * 256 + tid];
      sf[(size_t)b * 256 + tid] = a; }
}

__global__ __launch_bounds__(256) void time_embed_kernel(
    const int* __restrict__ timestep,
    const float* __restrict__ tp_w, const float* __restrict__ tp_b,
    const float* __restrict__ sf, float* __restrict__ tesf)
{
    int b = blockIdx.x, tid = threadIdx.x;
    __shared__ float emb[128];
    float t = (float)timestep[b];
    if (tid < 64) {
        float fr = __expf((float)tid * (-logf(10000.f) / 63.f));
        float ang = t * fr;
        emb[tid]      = __sinf(ang);
        emb[64 + tid] = __cosf(ang);
    }
    __syncthreads();
    float a = tp_b[tid];
    for (int k = 0; k < 128; ++k) a += emb[k] * tp_w[(size_t)k * 256 + tid];
    tesf[(size_t)b * 256 + tid] = a + sf[(size_t)b * 256 + tid];
}

// ===========================================================================
// Workgroup GEMM over a 64-row activation tile using V_WMMA_F32_16X16X32_F16.
// src/dst: LDS f16 row-major (A-fragment = 2x4 consecutive dwords -> b128).
// wp: fragment-major f16 weights (lane's B fragment = 8 contiguous dwords).
// 8 wave32s share 4 M-subtiles x (N/64) groups; each task: 1 A fragment feeds
// 4 WMMAs into 4 f32 accumulators (4-way N-blocking).
// ===========================================================================
template <int K, bool RELU, bool RESID, bool ADDVEC>
__device__ __forceinline__ void wg_gemm(
    const _Float16* __restrict__ src, int srcStride,
    _Float16* __restrict__ dst, int dstStride,
    const unsigned int* __restrict__ wp,
    const float* __restrict__ bias, const float* __restrict__ vec, int N)
{
    const int tid  = threadIdx.x;
    const int wv   = tid >> 5;
    const int lane = tid & 31;
    const int lhi  = lane >> 4;
    const int lm   = lane & 15;
    const int ntiles = N >> 4;
    const int ntasks = 4 * (N >> 6);      // 4 m-subtiles x groups of 4 n-tiles
    for (int task = wv; task < ntasks; task += 8) {
        const int m0  = (task & 3) << 4;
        const int nt0 = (task >> 2) << 2;   // first of 4 n-tiles
        v8f acc[4] = {};
        const unsigned int* arow =
            (const unsigned int*)(src + (size_t)(m0 + lm) * srcStride);
        #pragma unroll
        for (int kk = 0; kk < K; kk += 32) {
            Frag A;
            #pragma unroll
            for (int p = 0; p < 8; ++p) {   // p 0-3 and 4-7 each contiguous -> b128
                int k = kk + ((p >> 2) << 4) + (lhi << 3) + ((p & 3) << 1);
                A.u[p] = arow[k >> 1];
            }
            Frag B[4];
            #pragma unroll
            for (int j = 0; j < 4; ++j) {
                const unsigned int* bp = wp +
                    ((((size_t)(kk >> 5) * ntiles) + nt0 + j) << 8) + (lane << 3);
                #pragma unroll
                for (int q = 0; q < 8; ++q) B[j].u[q] = bp[q];   // 32B contiguous
            }
            #pragma unroll
            for (int j = 0; j < 4; ++j)
                acc[j] = __builtin_amdgcn_wmma_f32_16x16x32_f16(
                    false, A.v, false, B[j].v, (short)0, acc[j], false, false);
        }
        #pragma unroll
        for (int j = 0; j < 4; ++j) {
            int n = ((nt0 + j) << 4) + lm;
            float add = bias[n];
            if (ADDVEC) add += vec[n];
            #pragma unroll
            for (int r = 0; r < 8; ++r) {   // C/D: VGPR r -> M = m0 + 8*lhi + r
                int m = m0 + (lhi << 3) + r;
                float v = acc[j][r] + add;
                if (RESID) v += (float)dst[(size_t)m * dstStride + n];
                if (RELU)  v = fmaxf(v, 0.f);
                dst[(size_t)m * dstStride + n] = (_Float16)v;
            }
        }
    }
}

__global__ __launch_bounds__(256) void noise_main_kernel(
    const float* __restrict__ down,
    const unsigned int* __restrict__ wp_pe2, const unsigned int* __restrict__ wp_pe3,
    const unsigned int* __restrict__ wp_lw1, const unsigned int* __restrict__ wp_lw2,
    const unsigned int* __restrict__ wp_ow1, const unsigned int* __restrict__ wp_ow2,
    const float* __restrict__ pe_w1, const float* __restrict__ pe_b1,
    const float* __restrict__ pe_b2, const float* __restrict__ pe_b3,
    const float* __restrict__ lb1, const float* __restrict__ lb2,
    const float* __restrict__ ob1, const float* __restrict__ ob2,
    const float* __restrict__ ow3, const float* __restrict__ ob3,
    const float* __restrict__ tesf, float* __restrict__ pred)
{
    extern __shared__ _Float16 smem[];            // 100352 B (CDNA5: 320KB/WGP)
    _Float16* X = smem;                            // [64][XSTRIDE]
    _Float16* H = smem + MTILE * XSTRIDE;          // [64][HSTRIDE]
    const int tile = blockIdx.x, b = blockIdx.y, tid = threadIdx.x;
    const int base = tile * MTILE;

    // stage 0: points -> 3->128 (pe_w1, relu) into H
    {
        int m = tid >> 2, q = tid & 3;   // 64 rows x 4 feature groups of 32
        int mg = base + m;
        float px = 0.f, py = 0.f, pz = 0.f;
        if (mg < TARGET) {
            const float* p = down + ((size_t)b * TARGET + mg) * 3;
            px = p[0]; py = p[1]; pz = p[2];
        }
        #pragma unroll
        for (int f0 = 0; f0 < 32; ++f0) {
            int f = q * 32 + f0;
            float v = px * pe_w1[f] + py * pe_w1[128 + f] + pz * pe_w1[256 + f] + pe_b1[f];
            H[m * HSTRIDE + f] = (_Float16)fmaxf(v, 0.f);
        }
    }
    __syncthreads();
    wg_gemm<128, true,  false, false>(H, HSTRIDE, X, XSTRIDE, wp_pe2, pe_b2, nullptr, 256);
    __syncthreads();
    wg_gemm<256, false, false, true >(X, XSTRIDE, H, HSTRIDE, wp_pe3, pe_b3,
                                      tesf + (size_t)b * 256, 256);
    __syncthreads();
    { // copy H[:, :256] -> X (x now lives in X)
        int m = tid >> 2, q = tid & 3;
        const unsigned int* s = (const unsigned int*)(H + m * HSTRIDE + q * 64);
        unsigned int* d = (unsigned int*)(X + m * XSTRIDE + q * 64);
        #pragma unroll
        for (int i = 0; i < 32; ++i) d[i] = s[i];
    }
    __syncthreads();
    for (int blk = 0; blk < 6; ++blk) {   // 6 residual blocks 256->512->256
        wg_gemm<256, true,  false, false>(X, XSTRIDE, H, HSTRIDE,
                                          wp_lw1 + (size_t)blk * 65536,
                                          lb1 + blk * 512, nullptr, 512);
        __syncthreads();
        wg_gemm<512, false, true,  false>(H, HSTRIDE, X, XSTRIDE,
                                          wp_lw2 + (size_t)blk * 65536,
                                          lb2 + blk * 256, nullptr, 256);
        __syncthreads();
    }
    wg_gemm<256, true, false, false>(X, XSTRIDE, H, HSTRIDE, wp_ow1, ob1, nullptr, 256);
    __syncthreads();
    wg_gemm<256, true, false, false>(H, HSTRIDE, X, XSTRIDE, wp_ow2, ob2, nullptr, 128);
    __syncthreads();
    if (tid < MTILE) {   // final 128 -> 3 per point
        int mg = base + tid;
        if (mg < TARGET) {
            const _Float16* row = X + tid * XSTRIDE;
            float o0 = ob3[0], o1 = ob3[1], o2 = ob3[2];
            for (int k = 0; k < 128; ++k) {
                float xv = (float)row[k];
                o0 += xv * ow3[k * 3 + 0];
                o1 += xv * ow3[k * 3 + 1];
                o2 += xv * ow3[k * 3 + 2];
            }
            float* o = pred + ((size_t)b * TARGET + mg) * 3;
            o[0] = o0; o[1] = o1; o[2] = o2;
        }
    }
}

// ===========================================================================
extern "C" void kernel_launch(void* const* d_in, const int* in_sizes, int n_in,
                              void* d_out, int out_size, void* d_ws, size_t ws_size,
                              hipStream_t stream)
{
    (void)in_sizes; (void)n_in; (void)out_size; (void)ws_size;
    const float* noisy = (const float*)d_in[0];
    const int*   tstep = (const int*)d_in[1];
    const float* cond  = (const float*)d_in[2];
    const float* se_w1 = (const float*)d_in[3];  const float* se_b1 = (const float*)d_in[4];
    const float* se_w2 = (const float*)d_in[5];  const float* se_b2 = (const float*)d_in[6];
    const float* se_w3 = (const float*)d_in[7];  const float* se_b3 = (const float*)d_in[8];
    const float* sm_w1 = (const float*)d_in[9];  const float* sm_b1 = (const float*)d_in[10];
    const float* sm_w2 = (const float*)d_in[11]; const float* sm_b2 = (const float*)d_in[12];
    const float* pe_w1 = (const float*)d_in[13]; const float* pe_b1 = (const float*)d_in[14];
    const float* pe_w2 = (const float*)d_in[15]; const float* pe_b2 = (const float*)d_in[16];
    const float* pe_w3 = (const float*)d_in[17]; const float* pe_b3 = (const float*)d_in[18];
    const float* tp_w  = (const float*)d_in[19]; const float* tp_b  = (const float*)d_in[20];
    const float* sp_w  = (const float*)d_in[21]; const float* sp_b  = (const float*)d_in[22];
    const float* lw1   = (const float*)d_in[23]; const float* lb1   = (const float*)d_in[24];
    const float* lw2   = (const float*)d_in[25]; const float* lb2   = (const float*)d_in[26];
    const float* ow1   = (const float*)d_in[27]; const float* ob1   = (const float*)d_in[28];
    const float* ow2   = (const float*)d_in[29]; const float* ob2   = (const float*)d_in[30];
    const float* ow3   = (const float*)d_in[31]; const float* ob3   = (const float*)d_in[32];

    char* ws = (char*)d_ws;
    unsigned int* wp_pe2 = (unsigned int*)(ws + OFF_PE_W2);
    unsigned int* wp_pe3 = (unsigned int*)(ws + OFF_PE_W3);
    unsigned int* wp_lw1 = (unsigned int*)(ws + OFF_LW1);
    unsigned int* wp_lw2 = (unsigned int*)(ws + OFF_LW2);
    unsigned int* wp_ow1 = (unsigned int*)(ws + OFF_OW1);
    unsigned int* wp_ow2 = (unsigned int*)(ws + OFF_OW2);
    float* mmpart = (float*)(ws + OFF_MM_PART);
    float* mm     = (float*)(ws + OFF_MM);
    int*   hash   = (int*)(ws + OFF_HASH);
    int*   sel    = (int*)(ws + OFF_SEL);
    int*   bsum   = (int*)(ws + OFF_BSUM);
    int*   bexc   = (int*)(ws + OFF_BEXC);
    int*   tot    = (int*)(ws + OFF_TOT);
    int*   finalI = (int*)(ws + OFF_FINAL);
    float* down   = (float*)(ws + OFF_DOWN);
    float* gpart  = (float*)(ws + OFF_GPART);
    float* sf     = (float*)(ws + OFF_SF);
    float* tesf   = (float*)(ws + OFF_TESF);

    float* pred  = (float*)d_out;
    int*   idxO  = (int*)d_out + 2 * TARGET * 3;

    // 1. pack weights to f16 fragment-major
    auto pk = [&](const float* s, unsigned int* d, int K, int N) {
        int total = (K / 2) * N;
        pack_weights_kernel<<<(total + 255) / 256, 256, 0, stream>>>(s, d, K, N);
    };
    pk(pe_w2, wp_pe2, 128, 256);
    pk(pe_w3, wp_pe3, 256, 256);
    pk(lw1,   wp_lw1, 1536, 512);   // 6 stacked 256x512 (fragments never straddle)
    pk(lw2,   wp_lw2, 3072, 256);   // 6 stacked 512x256
    pk(ow1,   wp_ow1, 256, 256);
    pk(ow2,   wp_ow2, 256, 128);

    // 2. voxel downsample (4 = {noisy,cond} x {b0,b1})
    minmax_partial_kernel<<<dim3(64, 4), 256, 0, stream>>>(noisy, cond, mmpart);
    minmax_final_kernel<<<4, 64, 0, stream>>>(mmpart, mm);
    hash_init_kernel<<<dim3(TABLE / 256, 4), 256, 0, stream>>>(hash);
    hash_insert_kernel<<<dim3(NBLK_IN, 4), 256, 0, stream>>>(noisy, cond, mm, hash);
    mark_count_kernel<<<dim3(NBLK_IN, 4), 256, 0, stream>>>(noisy, cond, mm, hash, sel, bsum);
    scan_blocks_kernel<<<4, 32, 0, stream>>>(bsum, bexc, tot);
    scatter_kernel<<<dim3(NBLK_IN, 4), 256, 0, stream>>>(sel, bexc, tot, finalI);
    gather_kernel<<<dim3(NBLK_GATH, 4), 256, 0, stream>>>(noisy, cond, finalI, down, idxO);

    // 3. style encoder + style MLP
    style_point_kernel<<<dim3(NBLK_STY, 2), 256, 0, stream>>>(
        down, se_w1, se_b1, se_w2, se_b2, se_w3, se_b3, gpart);
    style_mlp_kernel<<<2, 256, 0, stream>>>(gpart, sm_w1, sm_b1, sm_w2, sm_b2,
                                            sp_w, sp_b, sf);

    // 4. time embedding (+ style projection fold-in)
    time_embed_kernel<<<2, 256, 0, stream>>>(tstep, tp_w, tp_b, sf, tesf);

    // 5. main fused WMMA pipeline (64-row tiles, dynamic LDS)
    noise_main_kernel<<<dim3(NTILES_M, 2), 256, MAIN_LDS_BYTES, stream>>>(
        down, wp_pe2, wp_pe3, wp_lw1, wp_lw2, wp_ow1, wp_ow2,
        pe_w1, pe_b1, pe_b2, pe_b3, lb1, lb2, ob1, ob2, ow3, ob3, tesf, pred);
}
